// CasualAttention_34574486732905
// MI455X (gfx1250) — matrix operand order repaired and checked by reading, not compiled, not run on hardware
//
#include <hip/hip_runtime.h>

typedef __attribute__((ext_vector_type(16))) _Float16 v16h;
typedef __attribute__((ext_vector_type(8)))  _Float16 v8h;
typedef __attribute__((ext_vector_type(4)))  _Float16 v4h;
typedef __attribute__((ext_vector_type(8)))  float    v8f;
typedef __attribute__((ext_vector_type(4)))  int      v4i;

#define BB 4
#define SS 4096
#define DIN 768
#define DH 64

#if defined(__has_builtin)
#  if __has_builtin(__builtin_amdgcn_global_load_async_to_lds_b128)
#    define HAVE_ASYNC_LDS 1
#  endif
#  if __has_builtin(__builtin_amdgcn_s_wait_asynccnt)
#    define HAVE_WAIT_ASYNC 1
#  endif
#endif

static __device__ __forceinline__ v8f wmma_f16(v16h a, v16h b, v8f c) {
    return __builtin_amdgcn_wmma_f32_16x16x32_f16(false, a, false, b, (short)0, c,
                                                  false, false);
}

#ifdef HAVE_ASYNC_LDS
typedef __attribute__((address_space(1))) v4i* gas_v4i_ptr;
typedef __attribute__((address_space(3))) v4i* las_v4i_ptr;

template <int OFF>
static __device__ __forceinline__ void async_cp16(const _Float16* g, _Float16* l) {
    __builtin_amdgcn_global_load_async_to_lds_b128(
        (gas_v4i_ptr)(void*)g, (las_v4i_ptr)(void*)l, OFF, 0);
}
#endif

static __device__ __forceinline__ void wait_async0() {
#ifdef HAVE_ASYNC_LDS
#  ifdef HAVE_WAIT_ASYNC
    __builtin_amdgcn_s_wait_asynccnt(0);
#  else
    asm volatile("s_wait_asynccnt 0" ::: "memory");
#  endif
#endif
}

// -------- Kernel 0: pack W (fp32 -> f16) into B-fragment-major layout --------
// wp[((kchunk*3 + mat)*64 + col)*32 + klocal] = W_mat[(kchunk*32+klocal)*64+col]
__global__ __launch_bounds__(256) void pack_w_kernel(
    const float* __restrict__ Wq, const float* __restrict__ Wk,
    const float* __restrict__ Wv, _Float16* __restrict__ wp)
{
    int idx = blockIdx.x * 256 + threadIdx.x;
    if (idx >= 3 * DIN * DH) return;
    int kl  = idx & 31;
    int col = (idx >> 5) & 63;
    int mat = (idx >> 11) % 3;
    int m   = idx / (3 * DH * 32);
    const float* W = (mat == 0) ? Wq : ((mat == 1) ? Wk : Wv);
    wp[idx] = (_Float16)W[(m * 32 + kl) * DH + col];
}

// -------- Kernel 1: QKV projection, 16 rows of x per 128-thread block --------
__global__ __launch_bounds__(128) void qkv_kernel(
    const float* __restrict__ x, const _Float16* __restrict__ wp,
    _Float16* __restrict__ q, _Float16* __restrict__ k, _Float16* __restrict__ v)
{
    __shared__ _Float16 xs[16 * DIN];          // 24 KB f16 tile of x
    const long row0 = (long)blockIdx.x * 16;
    const int  tid  = threadIdx.x;

    // vectorized fp32 -> f16 stage (float4 in, v4h out)
    for (int i = tid * 4; i < 16 * DIN; i += 128 * 4) {
        float4 f = *(const float4*)(x + row0 * DIN + i);
        v4h h; h[0] = (_Float16)f.x; h[1] = (_Float16)f.y;
               h[2] = (_Float16)f.z; h[3] = (_Float16)f.w;
        *(v4h*)&xs[i] = h;
    }
    __syncthreads();

    const int lane  = tid & 31;
    const int laneM = lane & 15;
    const int hi    = lane >> 4;
    const int n0    = (tid >> 5) * 16;         // wave -> N-tile

    v8f cq = {}, ck = {}, cv = {};
    for (int k0 = 0; k0 < DIN; k0 += 32) {
        const int m = k0 >> 5;
        // A fragment: two contiguous 8-half runs from LDS
        union { v16h v; v8h h[2]; } a;
        a.h[0] = *(const v8h*)&xs[laneM * DIN + k0 + hi * 16];
        a.h[1] = *(const v8h*)&xs[laneM * DIN + k0 + hi * 16 + 8];
        // B fragments: single contiguous v16h from packed W
        const int base = ((m * 3) * DH + n0 + laneM) * 32 + hi * 16;
        v16h bq = *(const v16h*)&wp[base];
        v16h bk = *(const v16h*)&wp[base + DH * 32];
        v16h bv = *(const v16h*)&wp[base + 2 * DH * 32];
        cq = wmma_f16(a.v, bq, cq);
        ck = wmma_f16(a.v, bk, ck);
        cv = wmma_f16(a.v, bv, cv);
    }

#pragma unroll
    for (int r = 0; r < 8; ++r) {
        long row = row0 + r + hi * 8;
        int  col = n0 + laneM;
        q[row * DH + col] = (_Float16)cq[r];
        k[row * DH + col] = (_Float16)ck[r];
        v[row * DH + col] = (_Float16)cv[r];
    }
}

// -------- Kernel 2: flash attention, one wave per 16-query tile -------------
__global__ __launch_bounds__(32) void attn_kernel(
    const _Float16* __restrict__ q, const _Float16* __restrict__ k,
    const _Float16* __restrict__ v, float* __restrict__ out)
{
    __shared__ _Float16 qs[16 * DH];     // query tile
    __shared__ _Float16 ks[32 * DH];     // key block, row-major
    __shared__ _Float16 vt[DH * 32];     // value block, TRANSPOSED [dim][key]
    __shared__ float    ps[16 * 32];     // probability tile for re-layout

    const int b  = blockIdx.x >> 8;
    const int q0 = (blockIdx.x & 255) * 16;

    const _Float16* qb = q + ((long)b * SS + q0) * DH;
    const _Float16* kb = k + (long)b * SS * DH;
    const _Float16* vb = v + (long)b * SS * DH;

    const int lane  = threadIdx.x;
    const int laneM = lane & 15;
    const int hi    = lane >> 4;

    for (int i = lane * 4; i < 16 * DH; i += 32 * 4)
        *(v4h*)&qs[i] = *(const v4h*)&qb[i];
    __syncthreads();

    // Q A-fragments, built once (two contiguous v8h ds loads each)
    union { v16h v; v8h h[2]; } aq0, aq1;
    aq0.h[0] = *(const v8h*)&qs[laneM * DH + hi * 16];
    aq0.h[1] = *(const v8h*)&qs[laneM * DH + hi * 16 + 8];
    aq1.h[0] = *(const v8h*)&qs[laneM * DH + 32 + hi * 16];
    aq1.h[1] = *(const v8h*)&qs[laneM * DH + 32 + hi * 16 + 8];

    float m[8], lsum[8];
#pragma unroll
    for (int r = 0; r < 8; ++r) { m[r] = -3.0e38f; lsum[r] = 0.0f; }
    v8f acc[4] = {{}, {}, {}, {}};

    const float scale = 0.125f;
    const int   kend  = q0 + 16;

    for (int kblk = 0; kblk < kend; kblk += 32) {
        __builtin_prefetch(&kb[(long)(kblk + 32) * DH], 0, 0);
        __builtin_prefetch(&vb[(long)(kblk + 32) * DH], 0, 0);

        // ---- stage K block (async-to-LDS when available) ------------------
        const _Float16* krow = kb + (long)(kblk + lane) * DH;
#ifdef HAVE_ASYNC_LDS
        {
            _Float16* lrow = &ks[lane * DH];
            async_cp16<0>(krow, lrow);   async_cp16<16>(krow, lrow);
            async_cp16<32>(krow, lrow);  async_cp16<48>(krow, lrow);
            async_cp16<64>(krow, lrow);  async_cp16<80>(krow, lrow);
            async_cp16<96>(krow, lrow);  async_cp16<112>(krow, lrow);
        }
#else
#pragma unroll
        for (int c = 0; c < 4; ++c)
            *(v16h*)&ks[lane * DH + c * 16] = *(const v16h*)(krow + c * 16);
#endif
        // ---- stage V block transposed: vt[dim][key] -----------------------
        const _Float16* vrow = vb + (long)(kblk + lane) * DH;
#pragma unroll
        for (int c = 0; c < 4; ++c) {
            v16h ch = *(const v16h*)(vrow + c * 16);
#pragma unroll
            for (int j = 0; j < 16; ++j)
                vt[(c * 16 + j) * 32 + lane] = ch[j];
        }
        wait_async0();
        __syncthreads();

        // ---- scores: contiguous v16h B-fragments from LDS -----------------
        v16h b0 = *(const v16h*)&ks[laneM * DH + hi * 16];
        v16h b1 = *(const v16h*)&ks[laneM * DH + 32 + hi * 16];
        v16h c0 = *(const v16h*)&ks[(16 + laneM) * DH + hi * 16];
        v16h c1 = *(const v16h*)&ks[(16 + laneM) * DH + 32 + hi * 16];
        v8f s1 = {}, s2 = {};
        s1 = wmma_f16(aq0.v, b0, s1);
        s1 = wmma_f16(aq1.v, b1, s1);
        s2 = wmma_f16(aq0.v, c0, s2);
        s2 = wmma_f16(aq1.v, c1, s2);

        // ---- online softmax ----------------------------------------------
#pragma unroll
        for (int r = 0; r < 8; ++r) {
            int   qrow = q0 + r + hi * 8;
            float x1 = (kblk + laneM <= qrow)      ? s1[r] * scale : -1.0e30f;
            float x2 = (kblk + 16 + laneM <= qrow) ? s2[r] * scale : -1.0e30f;

            float rm = fmaxf(x1, x2);
            rm = fmaxf(rm, __shfl_xor(rm, 1, 16));
            rm = fmaxf(rm, __shfl_xor(rm, 2, 16));
            rm = fmaxf(rm, __shfl_xor(rm, 4, 16));
            rm = fmaxf(rm, __shfl_xor(rm, 8, 16));

            float mnew = fmaxf(m[r], rm);
            float corr = __expf(m[r] - mnew);
            float p1   = __expf(x1 - mnew);
            float p2   = __expf(x2 - mnew);

            float rs = p1 + p2;
            rs += __shfl_xor(rs, 1, 16);
            rs += __shfl_xor(rs, 2, 16);
            rs += __shfl_xor(rs, 4, 16);
            rs += __shfl_xor(rs, 8, 16);

            lsum[r] = lsum[r] * corr + rs;
            m[r]    = mnew;
#pragma unroll
            for (int t = 0; t < 4; ++t) acc[t][r] *= corr;

            ps[(r + hi * 8) * 32 + laneM]      = p1;
            ps[(r + hi * 8) * 32 + 16 + laneM] = p2;
        }
        __syncthreads();

        // ---- P A-fragment + transposed-V B-fragments, acc += P.V ----------
        v16h ap;
#pragma unroll
        for (int h = 0; h < 16; ++h) {
            int kk = h + hi * 16;
            ap[h] = (_Float16)ps[laneM * 32 + kk];
        }
#pragma unroll
        for (int t = 0; t < 4; ++t) {
            v16h bvf = *(const v16h*)&vt[(t * 16 + laneM) * 32 + hi * 16];
            acc[t] = wmma_f16(ap, bvf, acc[t]);
        }
        __syncthreads();   // protect ks/vt/ps before next iteration's stage
    }

    // ---- epilogue ---------------------------------------------------------
#pragma unroll
    for (int r = 0; r < 8; ++r) {
        float inv = 1.0f / lsum[r];
        long  row = (long)b * SS + q0 + r + hi * 8;
#pragma unroll
        for (int t = 0; t < 4; ++t)
            out[row * DH + t * 16 + laneM] = acc[t][r] * inv;
    }
}

extern "C" void kernel_launch(void* const* d_in, const int* in_sizes, int n_in,
                              void* d_out, int out_size, void* d_ws, size_t ws_size,
                              hipStream_t stream) {
    const float* x  = (const float*)d_in[0];
    const float* Wq = (const float*)d_in[1];
    const float* Wk = (const float*)d_in[2];
    const float* Wv = (const float*)d_in[3];
    float* out = (float*)d_out;

    const size_t per = (size_t)BB * SS * DH;       // elements per Q/K/V buffer
    _Float16* qws = (_Float16*)d_ws;
    _Float16* kws = qws + per;
    _Float16* vws = kws + per;
    _Float16* wp  = vws + per;                     // packed weights (288 KB)

    pack_w_kernel<<<(3 * DIN * DH + 255) / 256, 256, 0, stream>>>(Wq, Wk, Wv, wp);
    qkv_kernel<<<BB * SS / 16, 128, 0, stream>>>(x, wp, qws, kws, vws);
    attn_kernel<<<BB * SS / 16, 32, 0, stream>>>(qws, kws, vws, out);
}
